// NeuralStatePredictor_70497593196714
// MI455X (gfx1250) — matrix operand order repaired and checked by reading, not compiled, not run on hardware
//
#include <hip/hip_runtime.h>
#include <hip/hip_bf16.h>
#include <math.h>

// ---------------------------------------------------------------------------
// Types for CDNA5 WMMA (gfx1250): 16x16x32 bf16 -> f32 accumulate
// ---------------------------------------------------------------------------
typedef __attribute__((ext_vector_type(16))) __bf16 v16bf;
typedef __attribute__((ext_vector_type(8)))  __bf16 v8bf;
typedef __attribute__((ext_vector_type(8)))  float  v8f;

#define BF(x) ((__bf16)(x))
// dst[b0..b0+3] = bf16(v.xyzw)  (b0 literal so vector subscripts fold)
#define CVT4(dst, b0, v)                                                      \
  do {                                                                        \
    (dst)[(b0) + 0] = BF((v).x);                                              \
    (dst)[(b0) + 1] = BF((v).y);                                              \
    (dst)[(b0) + 2] = BF((v).z);                                              \
    (dst)[(b0) + 3] = BF((v).w);                                              \
  } while (0)

// ---------------------------------------------------------------------------
// GEMM: Y[M,N] = act( X[M,K] @ W[N,K]^T + bias[N] )
//
// Block = 256 threads = 8 waves. Wave w owns rows [mbase+16w, +16) x the
// block's 64 output columns as four 16x16 WMMA tiles; K-loop step 32.
//
// B (weight) tile staging: the 64Nx32K fp32 tile (8KB) is loaded ONCE per
// block per k-step by all 256 threads cooperatively, converted to bf16, and
// stored to LDS pre-swizzled in WMMA B-fragment order:
//     ldsB[buf][nt][lane][16 bf16]   (lane = N col (lane&15), K-half (lane>>4))
// Double-buffered (2 x 4KB), one __syncthreads per k-step. Each wave then
// fetches a fragment with one 32B LDS read. B fragments are held in four
// DISTINCT variables so all 8 ds_loads issue before a single s_wait_dscnt and
// the 4 WMMAs run back-to-back (distinct accumulators -> no D->A/B hazard).
//
// A-fragment (16-bit A 16x32, ISA 7.12.2): lane half h=lane>>4, m=lane&15;
// element e holds K=(e<8?e:e+8)+8h -> two contiguous 8-float runs per lane;
// loaded straight from global (X is small and L2-resident).
// C/D layout: VGPR v holds M = v + 8h, N = lane&15.
//
// ACT: 0=none, 1=ReLU.  BOUNDED: guard N (only x_proj, N=40, needs it).
// ---------------------------------------------------------------------------
template <int ACT, bool BOUNDED>
__global__ __launch_bounds__(256) void gemm_wmma_bf16(
    const float* __restrict__ X, const float* __restrict__ W,
    const float* __restrict__ bias, float* __restrict__ Y,
    int M, int N, int K) {
  __shared__ alignas(64) v8bf ldsB[2][4][32][2];

  const int tid  = threadIdx.x;
  const int lane = tid & 31;
  const int wave = tid >> 5;
  const int h    = lane >> 4;
  const int l16  = lane & 15;
  const int mbase = blockIdx.y * 128 + wave * 16;
  const int nbase = blockIdx.x * 64;

  // ---- cooperative B-fill role for this thread ----
  const int fc    = tid >> 1;   // 0..127 : (nt, lane) chunk
  const int fpart = tid & 1;    // 0..1   : low/high 8 K-values of the fragment
  const int fnt   = fc >> 5;
  const int flane = fc & 31;
  const int fn    = nbase + fnt * 16 + (flane & 15);
  const int fh    = flane >> 4;
  const bool fvalid = !BOUNDED || (fn < N);
  const float* fwp0 = W + (size_t)fn * K + fh * 8 + fpart * 16;

  auto fill_regs = [&](int k0, float4& x0, float4& x1) {
    if (fvalid) {
      const float* wp = fwp0 + k0;
      __builtin_prefetch(wp + 64, 0, 1);  // global_prefetch_b8, 256B ahead
      x0 = *(const float4*)(wp + 0);
      x1 = *(const float4*)(wp + 4);
    } else {
      x0 = make_float4(0.f, 0.f, 0.f, 0.f);
      x1 = x0;
    }
  };
  auto store_lds = [&](int buf, const float4& x0, const float4& x1) {
    v8bf v;
    v[0] = BF(x0.x); v[1] = BF(x0.y); v[2] = BF(x0.z); v[3] = BF(x0.w);
    v[4] = BF(x1.x); v[5] = BF(x1.y); v[6] = BF(x1.z); v[7] = BF(x1.w);
    ldsB[buf][fnt][flane][fpart] = v;
  };

  // prologue: stage first B tile
  {
    float4 x0, x1;
    fill_regs(0, x0, x1);
    store_lds(0, x0, x1);
  }
  __syncthreads();

  v8f acc0 = {}, acc1 = {}, acc2 = {}, acc3 = {};
  const float* __restrict__ arowp = X + (size_t)(mbase + l16) * K;

  int p = 0;
  for (int k0 = 0; k0 < K; k0 += 32) {
    // ---- A fragment (global, per-wave unique) ----
    const float* ap = arowp + k0 + h * 8;
    float4 a0 = *(const float4*)(ap + 0);
    float4 a1 = *(const float4*)(ap + 4);
    float4 a2 = *(const float4*)(ap + 16);
    float4 a3 = *(const float4*)(ap + 20);
    v16bf afrag;
    CVT4(afrag, 0, a0);
    CVT4(afrag, 4, a1);
    CVT4(afrag, 8, a2);
    CVT4(afrag, 12, a3);

    // ---- kick off next B tile load into registers ----
    const bool have_next = (k0 + 32 < K);
    float4 x0, x1;
    if (have_next) fill_regs(k0 + 32, x0, x1);

    // ---- fetch all 4 B fragments from LDS (distinct regs) ----
    v16bf bf0 = *(const v16bf*)&ldsB[p][0][lane][0];
    v16bf bf1 = *(const v16bf*)&ldsB[p][1][lane][0];
    v16bf bf2 = *(const v16bf*)&ldsB[p][2][lane][0];
    v16bf bf3 = *(const v16bf*)&ldsB[p][3][lane][0];

    // ---- stage next tile into the other buffer (independent of WMMAs) ----
    if (have_next) store_lds(p ^ 1, x0, x1);

    // ---- 4 back-to-back WMMAs ----
    acc0 = __builtin_amdgcn_wmma_f32_16x16x32_bf16(false, afrag, false, bf0,
                                                   (short)0, acc0, false, false);
    acc1 = __builtin_amdgcn_wmma_f32_16x16x32_bf16(false, afrag, false, bf1,
                                                   (short)0, acc1, false, false);
    acc2 = __builtin_amdgcn_wmma_f32_16x16x32_bf16(false, afrag, false, bf2,
                                                   (short)0, acc2, false, false);
    acc3 = __builtin_amdgcn_wmma_f32_16x16x32_bf16(false, afrag, false, bf3,
                                                   (short)0, acc3, false, false);

    __syncthreads();
    p ^= 1;
  }

  // ---- epilogue: bias + activation + store (C layout: M = v + 8h) ----
  v8f accs[4] = {acc0, acc1, acc2, acc3};
#pragma unroll
  for (int nt = 0; nt < 4; ++nt) {
    const int gcol = nbase + nt * 16 + l16;
    if (BOUNDED && gcol >= N) continue;
    const float bv = bias ? bias[gcol] : 0.0f;
#pragma unroll
    for (int v = 0; v < 8; ++v) {
      const int grow = mbase + h * 8 + v;
      float val = accs[nt][v] + bv;
      if (ACT == 1) val = fmaxf(val, 0.0f);
      Y[(size_t)grow * N + gcol] = val;
    }
  }
}

// ---------------------------------------------------------------------------
// concat: xcat[b, 0:8192] = state, xcat[b, 8192:8448] = action
// ---------------------------------------------------------------------------
__global__ void concat_kernel(const float* __restrict__ state,
                              const float* __restrict__ action,
                              float* __restrict__ out, int total) {
  int i = blockIdx.x * blockDim.x + threadIdx.x;
  if (i >= total) return;
  int b = i / 8448, c = i % 8448;
  out[i] = (c < 8192) ? state[(size_t)b * 8192 + c]
                      : action[(size_t)b * 256 + (c - 8192)];
}

// ---------------------------------------------------------------------------
// causal depthwise conv1d (D_CONV=4) + bias + SiLU
// ---------------------------------------------------------------------------
__global__ void conv_silu_kernel(const float* __restrict__ xr,
                                 const float* __restrict__ cw,
                                 const float* __restrict__ cb,
                                 float* __restrict__ xc, int total) {
  int i = blockIdx.x * blockDim.x + threadIdx.x;
  if (i >= total) return;
  int c = i & 255;
  int t = i >> 8;
  int l = t & 63;
  float acc = cb[c];
#pragma unroll
  for (int j = 0; j < 4; ++j) {
    int lj = l - 3 + j;
    if (lj >= 0) acc += xr[(size_t)(t - 3 + j) * 512 + c] * cw[c * 4 + j];
  }
  xc[i] = acc / (1.0f + __expf(-acc));  // silu
}

// ---------------------------------------------------------------------------
// delta = softplus( x_dbl[:, :8] @ dt_w^T + dt_b )   (K=8: plain VALU kernel)
// ---------------------------------------------------------------------------
__global__ void dt_softplus_kernel(const float* __restrict__ xdbl,
                                   const float* __restrict__ dtw,
                                   const float* __restrict__ dtb,
                                   float* __restrict__ delta, int total) {
  int i = blockIdx.x * blockDim.x + threadIdx.x;
  if (i >= total) return;
  int d = i & 255;
  int t = i >> 8;
  float s = dtb[d];
  const float* xp = xdbl + (size_t)t * 40;
  const float* wp = dtw + d * 8;
#pragma unroll
  for (int r = 0; r < 8; ++r) s += xp[r] * wp[r];
  delta[i] = (s > 20.0f) ? s : log1pf(__expf(s));
}

// ---------------------------------------------------------------------------
// selective scan over l=0..63, one thread per (batch b, channel d).
// ---------------------------------------------------------------------------
__global__ void scan_kernel(const float* __restrict__ delta,
                            const float* __restrict__ xc,
                            const float* __restrict__ xdbl,
                            const float* __restrict__ xr,
                            const float* __restrict__ Alog,
                            const float* __restrict__ Dskip,
                            float* __restrict__ y, int total) {
  int i = blockIdx.x * blockDim.x + threadIdx.x;
  if (i >= total) return;  // total = 128*256
  int d = i & 255;
  int b = i >> 8;
  float An[16];
#pragma unroll
  for (int n = 0; n < 16; ++n) An[n] = -__expf(Alog[d * 16 + n]);
  const float Dk = Dskip[d];
  float hh[16];
#pragma unroll
  for (int n = 0; n < 16; ++n) hh[n] = 0.0f;

  for (int l = 0; l < 64; ++l) {
    int t = b * 64 + l;
    float dt = delta[(size_t)t * 256 + d];
    float u  = xc[(size_t)t * 256 + d];
    const float* bc = xdbl + (size_t)t * 40;
    float yt = 0.0f;
#pragma unroll
    for (int n = 0; n < 16; ++n) {
      hh[n] = __expf(dt * An[n]) * hh[n] + dt * bc[8 + n] * u;
      yt += hh[n] * bc[24 + n];
    }
    yt += u * Dk;
    float r = xr[(size_t)t * 512 + 256 + d];
    yt *= r / (1.0f + __expf(-r));  // * silu(res)
    y[(size_t)t * 256 + d] = yt;
  }
}

// ---------------------------------------------------------------------------
// host orchestration
// ---------------------------------------------------------------------------
extern "C" void kernel_launch(void* const* d_in, const int* in_sizes, int n_in,
                              void* d_out, int out_size, void* d_ws,
                              size_t ws_size, hipStream_t stream) {
  (void)in_sizes; (void)n_in; (void)out_size; (void)ws_size;

  const float* state    = (const float*)d_in[0];
  const float* action   = (const float*)d_in[1];
  const float* w0a      = (const float*)d_in[2];
  const float* b0a      = (const float*)d_in[3];
  const float* w0b      = (const float*)d_in[4];
  const float* b0b      = (const float*)d_in[5];
  const float* in_proj  = (const float*)d_in[6];
  const float* conv_w   = (const float*)d_in[7];
  const float* conv_b   = (const float*)d_in[8];
  const float* xproj_w  = (const float*)d_in[9];
  const float* dt_w     = (const float*)d_in[10];
  const float* dt_b     = (const float*)d_in[11];
  const float* A_log    = (const float*)d_in[12];
  const float* Dsk      = (const float*)d_in[13];
  const float* out_proj = (const float*)d_in[14];
  const float* w1a      = (const float*)d_in[15];
  const float* b1a      = (const float*)d_in[16];
  const float* w1b      = (const float*)d_in[17];
  const float* b1b      = (const float*)d_in[18];

  float* ws   = (float*)d_ws;
  float* xcat = ws; ws += (size_t)128 * 8448;
  float* hh0  = ws; ws += (size_t)128 * 8448;
  float* xA   = ws; ws += (size_t)128 * 8192;
  float* xB   = ws; ws += (size_t)128 * 8192;
  float* xr   = ws; ws += (size_t)8192 * 512;
  float* xc   = ws; ws += (size_t)8192 * 256;
  float* xdbl = ws; ws += (size_t)8192 * 40;
  float* dlt  = ws; ws += (size_t)8192 * 256;
  float* yb   = ws; ws += (size_t)8192 * 256;
  float* hh1  = ws; ws += (size_t)128 * 8192;

  auto gemm = [&](int act, const float* X, const float* W, const float* B,
                  float* Y, int M, int N, int K) {
    dim3 g((unsigned)((N + 63) / 64), (unsigned)(M / 128));
    dim3 blk(256);
    if ((N & 63) == 0) {
      if (act)
        gemm_wmma_bf16<1, false><<<g, blk, 0, stream>>>(X, W, B, Y, M, N, K);
      else
        gemm_wmma_bf16<0, false><<<g, blk, 0, stream>>>(X, W, B, Y, M, N, K);
    } else {
      if (act)
        gemm_wmma_bf16<1, true><<<g, blk, 0, stream>>>(X, W, B, Y, M, N, K);
      else
        gemm_wmma_bf16<0, true><<<g, blk, 0, stream>>>(X, W, B, Y, M, N, K);
    }
  };

  {
    int total = 128 * 8448;
    concat_kernel<<<(total + 255) / 256, 256, 0, stream>>>(state, action, xcat,
                                                           total);
  }

  // MLP0
  gemm(1, xcat, w0a, b0a, hh0, 128, 8448, 8448);
  gemm(0, hh0,  w0b, b0b, xA,  128, 8192, 8448);

  // 4 Mamba layers; x viewed as (T=8192 tokens, 128)
  for (int i = 0; i < 4; ++i) {
    float* xin  = (i & 1) ? xB : xA;
    float* xout = (i & 1) ? xA : xB;

    gemm(0, xin, in_proj + (size_t)i * 512 * 128, nullptr, xr, 8192, 512, 128);
    {
      int total = 8192 * 256;
      conv_silu_kernel<<<(total + 255) / 256, 256, 0, stream>>>(
          xr, conv_w + i * 1024, conv_b + i * 256, xc, total);
    }
    gemm(0, xc, xproj_w + (size_t)i * 40 * 256, nullptr, xdbl, 8192, 40, 256);
    {
      int total = 8192 * 256;
      dt_softplus_kernel<<<(total + 255) / 256, 256, 0, stream>>>(
          xdbl, dt_w + i * 2048, dt_b + i * 256, dlt, total);
    }
    {
      int total = 128 * 256;
      scan_kernel<<<(total + 255) / 256, 256, 0, stream>>>(
          dlt, xc, xdbl, xr, A_log + i * 4096, Dsk + i * 256, yb, total);
    }
    gemm(0, yb, out_proj + (size_t)i * 128 * 256, nullptr, xout, 8192, 128,
         256);
  }

  // after layer 3, result is in xA
  gemm(1, xA,  w1a, b1a, hh1, 128, 8192, 8192);
  gemm(0, hh1, w1b, b1b, (float*)d_out, 128, 8192, 8192);
}
